// CrossAttention_IPA_26542897889411
// MI455X (gfx1250) — matrix operand order repaired and checked by reading, not compiled, not run on hardware
//
#include <hip/hip_runtime.h>
#include <hip/hip_bf16.h>

typedef unsigned short u16;
typedef __attribute__((ext_vector_type(16))) __bf16 v16bf;
typedef __attribute__((ext_vector_type(8)))  float  v8f;

struct U16x16 { uint4 a, b; };

static __device__ __forceinline__ u16 f2bf(float f) {
  unsigned u = __builtin_bit_cast(unsigned, f);
  u += 0x7fffu + ((u >> 16) & 1u);          // round-to-nearest-even
  return (u16)(u >> 16);
}

static __device__ __forceinline__ v16bf frag16(uint4 lo, uint4 hi) {
  U16x16 t{lo, hi};
  return __builtin_bit_cast(v16bf, t);
}

// ---- cross-lane helpers for 16-lane butterfly reductions -------------------
// xor1/xor2 via DPP quad_perm (VALU, no wait); xor4/xor8 via ds_swizzle imm.
static __device__ __forceinline__ float lane_xor1(float v) {
  int i = __builtin_bit_cast(int, v);
  return __builtin_bit_cast(float, __builtin_amdgcn_update_dpp(i, i, 0xB1, 0xF, 0xF, true));
}
static __device__ __forceinline__ float lane_xor2(float v) {
  int i = __builtin_bit_cast(int, v);
  return __builtin_bit_cast(float, __builtin_amdgcn_update_dpp(i, i, 0x4E, 0xF, 0xF, true));
}
static __device__ __forceinline__ float lane_xor4(float v) {
  return __builtin_bit_cast(float,
      __builtin_amdgcn_ds_swizzle(__builtin_bit_cast(int, v), 0x101F));
}
static __device__ __forceinline__ float lane_xor8(float v) {
  return __builtin_bit_cast(float,
      __builtin_amdgcn_ds_swizzle(__builtin_bit_cast(int, v), 0x201F));
}
static __device__ __forceinline__ float red16_max(float v) {
  v = fmaxf(v, lane_xor1(v));
  v = fmaxf(v, lane_xor2(v));
  v = fmaxf(v, lane_xor4(v));
  v = fmaxf(v, lane_xor8(v));
  return v;
}
static __device__ __forceinline__ float red16_sum(float v) {
  v += lane_xor1(v);
  v += lane_xor2(v);
  v += lane_xor4(v);
  v += lane_xor8(v);
  return v;
}

// ---- conversion kernels ----------------------------------------------------
__global__ void cvt_f32_bf16(const float* __restrict__ in, u16* __restrict__ out, long n) {
  long i = (long)blockIdx.x * blockDim.x + threadIdx.x;
  long st = (long)gridDim.x * blockDim.x;
  for (; i < n; i += st) out[i] = f2bf(in[i]);
}

// W[K][N] f32 -> WT[N][K] bf16 (weights become K-contiguous per output column,
// so WMMA B fragments stream straight from global/L2; runs once).
__global__ void cvt_transpose_bf16(const float* __restrict__ w, u16* __restrict__ wt,
                                   int K, int N) {
  __shared__ float tile[32][33];
  const int k0 = blockIdx.x * 32, n0 = blockIdx.y * 32;
  const int tx = threadIdx.x, ty = threadIdx.y;   // (32, 8)
  #pragma unroll
  for (int j = 0; j < 4; ++j)
    tile[ty + j * 8][tx] = w[(size_t)(k0 + ty + j * 8) * N + n0 + tx];
  __syncthreads();
  #pragma unroll
  for (int j = 0; j < 4; ++j)
    wt[(size_t)(n0 + ty + j * 8) * K + k0 + tx] = f2bf(tile[tx][ty + j * 8]);
}

// ---- GEMM: C = A(bf16,[M][K]) * WT^T (WT is [N][K]) + bias -----------------
// No LDS, no barriers. Register-blocked: each wave owns 32 rows x 64 cols
// (two 16-row subtiles), so every B fragment feeds TWO wmmas:
// 12 b128 loads per 8 wmmas per k-step.
// MODE 0: Q path  : LN(per 64-col head) + RoPE + 1/sqrt(D) prescale, bf16 -> out0
// MODE 1: KV path : cols<splitN -> LN -> out0 row-major;
//                   cols>=splitN -> plain -> out1 TRANSPOSED [b][d'][s] (k-stride vStride)
// MODE 2: out path: f32 -> outf
template<int MODE>
__global__ void gemm_bf16(const u16* __restrict__ A, const u16* __restrict__ WT,
                          const float* __restrict__ bias,
                          u16* __restrict__ out0, u16* __restrict__ out1,
                          float* __restrict__ outf,
                          const float* __restrict__ lnw, const float* __restrict__ lnb,
                          const float* __restrict__ cosb, const float* __restrict__ sinb,
                          int M, int K, int N, int splitN, int rowsPerB, int vStride) {
  const int tid = threadIdx.x;        // 128 threads = 4 waves
  const int lane = tid & 31;
  const int wv = tid >> 5;            // 0..3 -> 32 rows each
  const int ln = lane & 15;
  const int hh = lane >> 4;
  const int m0 = blockIdx.x * 128;
  const int n0 = blockIdx.y * 64;
  const int mbase = m0 + wv * 32;

  v8f acc[2][4] = {};
  const u16* aRow0 = A + (size_t)(mbase + ln) * K;
  const u16* aRow1 = A + (size_t)(mbase + 16 + ln) * K;

  #pragma unroll 2
  for (int k0 = 0; k0 < K; k0 += 32) {
    // A frags: lanes 0-15 K={0..7,16..23}, lanes 16-31 K={8..15,24..31}
    v16bf aF0 = frag16(*(const uint4*)(aRow0 + k0 + hh * 8),
                       *(const uint4*)(aRow0 + k0 + 16 + hh * 8));
    v16bf aF1 = frag16(*(const uint4*)(aRow1 + k0 + hh * 8),
                       *(const uint4*)(aRow1 + k0 + 16 + hh * 8));
    if (k0 + 32 < K) {
      __builtin_prefetch(aRow0 + k0 + 32, 0, 1);
      __builtin_prefetch(aRow1 + k0 + 32, 0, 1);
    }
    #pragma unroll
    for (int t = 0; t < 4; ++t) {
      // B frag: lane N = ln, lanes 0-15 K={0..7,16..23}, lanes 16-31 K={8..15,24..31}; reused by both subtiles
      const u16* bp = WT + (size_t)(n0 + t * 16 + ln) * K + k0 + hh * 8;
      v16bf bF = frag16(*(const uint4*)bp, *(const uint4*)(bp + 16));
      acc[0][t] = __builtin_amdgcn_wmma_f32_16x16x32_bf16(false, aF0, false, bF,
                                                          (short)0, acc[0][t], false, false);
      acc[1][t] = __builtin_amdgcn_wmma_f32_16x16x32_bf16(false, aF1, false, bF,
                                                          (short)0, acc[1][t], false, false);
    }
  }

  // epilogue per subtile: rows m = mbase + mi*16 + hh*8 + r ; cols n = n0 + t*16 + ln
  #pragma unroll
  for (int mi = 0; mi < 2; ++mi) {
    float h[4][8];
    #pragma unroll
    for (int t = 0; t < 4; ++t) {
      float bv = bias[n0 + t * 16 + ln];
      #pragma unroll
      for (int r = 0; r < 8; ++r) h[t][r] = acc[mi][t][r] + bv;
    }
    const bool doLN = (MODE == 0) || (MODE == 1 && n0 < splitN);
    if (doLN) {
      #pragma unroll
      for (int r = 0; r < 8; ++r) {
        float s = h[0][r] + h[1][r] + h[2][r] + h[3][r];
        float mu = red16_sum(s) * (1.0f / 64.0f);
        float sq = 0.f;
        #pragma unroll
        for (int t = 0; t < 4; ++t) { float d = h[t][r] - mu; sq += d * d; }
        float var = red16_sum(sq) * (1.0f / 64.0f);
        float rstd = rsqrtf(var + 1e-6f);
        #pragma unroll
        for (int t = 0; t < 4; ++t) {
          int d = t * 16 + ln;
          h[t][r] = (h[t][r] - mu) * rstd * lnw[d] + lnb[d];
        }
      }
    }
    if (MODE == 0) { // RoPE (pair partner in adjacent lane) + fold attention scale into Q
      #pragma unroll
      for (int r = 0; r < 8; ++r) {
        int m = mbase + mi * 16 + hh * 8 + r;
        int s1 = m & 4095;   // S1 = 4096
        #pragma unroll
        for (int t = 0; t < 4; ++t) {
          int d = t * 16 + ln;
          int i = d >> 1;
          float c = cosb[s1 * 32 + i];
          float s = sinb[s1 * 32 + i];
          float v = h[t][r];
          float p = lane_xor1(v);
          float ro = ((ln & 1) == 0) ? (v * c - p * s) : (p * s + v * c);
          h[t][r] = ro * 0.125f;        // 1/sqrt(64) baked into stored Q
        }
      }
    }
    #pragma unroll
    for (int r = 0; r < 8; ++r) {
      const int m = mbase + mi * 16 + hh * 8 + r;
      const int bb = (MODE == 1) ? (m / rowsPerB) : 0;
      const int ss = (MODE == 1) ? (m % rowsPerB) : 0;
      #pragma unroll
      for (int t = 0; t < 4; ++t) {
        int nc = n0 + t * 16 + ln;
        if (MODE == 2) {
          outf[(size_t)m * N + nc] = h[t][r];
        } else if (MODE == 1) {
          if (nc < splitN) out0[(size_t)m * splitN + nc] = f2bf(h[t][r]);
          else out1[((size_t)bb * 1024 + (nc - splitN)) * vStride + ss] = f2bf(h[t][r]);
        } else {
          out0[(size_t)m * N + nc] = f2bf(h[t][r]);
        }
      }
    }
  }
}

// ---- fused dual flash attention: 4 independent waves/block -----------------
// K frags stream from global (row-major); V frags stream from pre-transposed
// global vT [b][h][d][s]. Only LDS use: the wave-private probs C->A layout
// shuffle. DS ops of one wave execute in order, so a compiler-level
// wave_barrier is sufficient -- no block barriers at all.
__global__ void attn_kernel(const u16* __restrict__ qb, const u16* __restrict__ kb,
                            const u16* __restrict__ vt, const u16* __restrict__ ipkb,
                            const u16* __restrict__ ipvt, u16* __restrict__ ctxb) {
  const int S1 = 4096, S2 = 256, SIP = 16, HID = 1024;
  __shared__ u16 Pw[4][16 * 32];
  const int tid = threadIdx.x;   // 128 threads
  const int lane = tid & 31;
  const int wv = tid >> 5;
  const int ln = lane & 15;
  const int hh = lane >> 4;
  const int b = blockIdx.z, h = blockIdx.y;
  const int q0 = blockIdx.x * 64 + wv * 16;

  // Q fragments (16 q rows x 64 d, pre-scaled by 1/8), reused for all chunks
  v16bf qA[2];
  {
    const u16* qp = qb + ((size_t)b * S1 + q0 + ln) * HID + h * 64;
    #pragma unroll
    for (int db = 0; db < 2; ++db)
      qA[db] = frag16(*(const uint4*)(qp + db * 32 + hh * 8),
                      *(const uint4*)(qp + db * 32 + 16 + hh * 8));
  }

  const u16* vbase = vt + ((size_t)b * 1024 + h * 64) * S2;     // [d][s], stride S2
  const u16* ipvbase = ipvt + ((size_t)b * 1024 + h * 64) * 32; // [d][k], stride 32 (zero-padded)

  float mrun[8], lrun[8];
  v8f O[4] = {};
  #pragma unroll
  for (int r = 0; r < 8; ++r) { mrun[r] = -3.0e38f; lrun[r] = 0.f; }

  for (int c = 0; c < 8; ++c) {
    const int k0 = c * 32;
    v8f S[2] = {};
    #pragma unroll
    for (int kt = 0; kt < 2; ++kt) {
      #pragma unroll
      for (int db = 0; db < 2; ++db) {
        const u16* kp = kb + ((size_t)b * S2 + k0 + kt * 16 + ln) * HID
                           + h * 64 + db * 32 + hh * 8;
        v16bf bK = frag16(*(const uint4*)kp, *(const uint4*)(kp + 16));
        S[kt] = __builtin_amdgcn_wmma_f32_16x16x32_bf16(false, qA[db], false, bK,
                                                        (short)0, S[kt], false, false);
      }
    }
    // online softmax over this 32-key chunk (scale already folded into Q)
    #pragma unroll
    for (int r = 0; r < 8; ++r) {
      float s0 = S[0][r], s1 = S[1][r];
      float mx = red16_max(fmaxf(s0, s1));
      float mnew = fmaxf(mrun[r], mx);
      float al = __expf(mrun[r] - mnew);
      float p0 = __expf(s0 - mnew), p1 = __expf(s1 - mnew);
      float rs = red16_sum(p0 + p1);
      lrun[r] = lrun[r] * al + rs;
      mrun[r] = mnew;
      #pragma unroll
      for (int t = 0; t < 4; ++t) O[t][r] *= al;
      const int m = hh * 8 + r;
      Pw[wv][m * 32 + ln] = f2bf(p0);
      Pw[wv][m * 32 + 16 + ln] = f2bf(p1);
    }
    __builtin_amdgcn_wave_barrier();   // order P stores before cross-lane P loads
    v16bf aP;
    {
      const u16* pp = Pw[wv] + ln * 32;
      aP = frag16(*(const uint4*)(pp + hh * 8), *(const uint4*)(pp + 16 + hh * 8));
    }
    __builtin_amdgcn_wave_barrier();   // order P loads before next chunk's stores
    #pragma unroll
    for (int t = 0; t < 4; ++t) {
      const u16* vp = vbase + (size_t)(t * 16 + ln) * S2 + k0 + hh * 8;
      v16bf bV = frag16(*(const uint4*)vp, *(const uint4*)(vp + 16));
      O[t] = __builtin_amdgcn_wmma_f32_16x16x32_bf16(false, aP, false, bV,
                                                     (short)0, O[t], false, false);
    }
  }

  // normalize main attention
  v8f Of[4];
  #pragma unroll
  for (int t = 0; t < 4; ++t)
    #pragma unroll
    for (int r = 0; r < 8; ++r) Of[t][r] = O[t][r] / lrun[r];

  // IP attention: one 16-key chunk (probs zero-padded to 32)
  v8f S0 = {};
  #pragma unroll
  for (int db = 0; db < 2; ++db) {
    const u16* kp = ipkb + ((size_t)b * SIP + ln) * HID + h * 64 + db * 32 + hh * 8;
    v16bf bK = frag16(*(const uint4*)kp, *(const uint4*)(kp + 16));
    S0 = __builtin_amdgcn_wmma_f32_16x16x32_bf16(false, qA[db], false, bK,
                                                 (short)0, S0, false, false);
  }
  float l2[8];
  #pragma unroll
  for (int r = 0; r < 8; ++r) {
    float s0 = S0[r];
    float mx = red16_max(s0);
    float p0 = __expf(s0 - mx);
    l2[r] = red16_sum(p0);
    const int m = hh * 8 + r;
    Pw[wv][m * 32 + ln] = f2bf(p0);
    Pw[wv][m * 32 + 16 + ln] = (u16)0;   // padded keys: zero prob
  }
  __builtin_amdgcn_wave_barrier();
  v8f O2[4] = {};
  v16bf aP2;
  {
    const u16* pp = Pw[wv] + ln * 32;
    aP2 = frag16(*(const uint4*)(pp + hh * 8), *(const uint4*)(pp + 16 + hh * 8));
  }
  __builtin_amdgcn_wave_barrier();
  #pragma unroll
  for (int t = 0; t < 4; ++t) {
    const u16* vp = ipvbase + (size_t)(t * 16 + ln) * 32 + hh * 8;
    v16bf bV = frag16(*(const uint4*)vp, *(const uint4*)(vp + 16));
    O2[t] = __builtin_amdgcn_wmma_f32_16x16x32_bf16(false, aP2, false, bV,
                                                    (short)0, O2[t], false, false);
  }
  // combine (ORG_RATIO = IP_RATIO = 1.0), store context bf16
  #pragma unroll
  for (int t = 0; t < 4; ++t) {
    #pragma unroll
    for (int r = 0; r < 8; ++r) {
      float v = Of[t][r] + O2[t][r] / l2[r];
      const int m = q0 + hh * 8 + r;
      const int col = h * 64 + t * 16 + ln;
      ctxb[((size_t)b * S1 + m) * HID + col] = f2bf(v);
    }
  }
}

extern "C" void kernel_launch(void* const* d_in, const int* in_sizes, int n_in,
                              void* d_out, int out_size, void* d_ws, size_t ws_size,
                              hipStream_t stream) {
  (void)in_sizes; (void)n_in; (void)out_size; (void)ws_size;
  const float* x     = (const float*)d_in[0];
  const float* y     = (const float*)d_in[1];
  const float* ip_y  = (const float*)d_in[2];
  const float* wq    = (const float*)d_in[3];
  const float* bq    = (const float*)d_in[4];
  const float* wkv   = (const float*)d_in[5];
  const float* bkv   = (const float*)d_in[6];
  const float* wip   = (const float*)d_in[7];
  const float* bip   = (const float*)d_in[8];
  const float* qn_w  = (const float*)d_in[9];
  const float* qn_b  = (const float*)d_in[10];
  const float* kn_w  = (const float*)d_in[11];
  const float* kn_b  = (const float*)d_in[12];
  const float* ipknw = (const float*)d_in[13];
  const float* ipknb = (const float*)d_in[14];
  const float* wo    = (const float*)d_in[15];
  const float* bo    = (const float*)d_in[16];
  const float* fcos  = (const float*)d_in[17];
  const float* fsin  = (const float*)d_in[18];
  float* out = (float*)d_out;

  const size_t M1 = 32768, HID = 1024, MKV = 2048, MIP = 128;
  u16* p = (u16*)d_ws;
  u16* xb    = p; p += M1 * HID;
  u16* qb    = p; p += M1 * HID;
  u16* ctxb  = p; p += M1 * HID;
  u16* yb    = p; p += MKV * HID;
  u16* kbuf  = p; p += MKV * HID;
  u16* vT    = p; p += MKV * HID;          // [b][h*64+d][s2]
  u16* ipyb  = p; p += MIP * HID;
  u16* ipkbf = p; p += MIP * HID;
  u16* ipvT  = p; p += (size_t)8 * 1024 * 32;  // [b][h*64+d][32], zero-padded
  u16* wqT   = p; p += HID * HID;
  u16* wkvT  = p; p += HID * 2048;
  u16* wipT  = p; p += HID * 2048;
  u16* woT   = p; p += HID * HID;

  auto cvt = [&](const float* src, u16* dst, long n) {
    int blocks = (int)((n + 255) / 256);
    if (blocks > 4096) blocks = 4096;
    cvt_f32_bf16<<<blocks, 256, 0, stream>>>(src, dst, n);
  };
  cvt(x, xb, (long)(M1 * HID));
  cvt(y, yb, (long)(MKV * HID));
  cvt(ip_y, ipyb, (long)(MIP * HID));

  // transpose+convert weights once: WT[n][k]
  cvt_transpose_bf16<<<dim3(1024 / 32, 1024 / 32), dim3(32, 8), 0, stream>>>(wq, wqT, 1024, 1024);
  cvt_transpose_bf16<<<dim3(1024 / 32, 2048 / 32), dim3(32, 8), 0, stream>>>(wkv, wkvT, 1024, 2048);
  cvt_transpose_bf16<<<dim3(1024 / 32, 2048 / 32), dim3(32, 8), 0, stream>>>(wip, wipT, 1024, 2048);
  cvt_transpose_bf16<<<dim3(1024 / 32, 1024 / 32), dim3(32, 8), 0, stream>>>(wo, woT, 1024, 1024);

  // zero the padded transposed ip-V buffer
  hipMemsetAsync(ipvT, 0, (size_t)8 * 1024 * 32 * sizeof(u16), stream);

  // Q projection + LN + RoPE (+1/8 prescale)
  gemm_bf16<0><<<dim3((unsigned)(M1 / 128), (unsigned)(HID / 64)), 128, 0, stream>>>(
      xb, wqT, bq, qb, nullptr, nullptr, qn_w, qn_b, fcos, fsin,
      (int)M1, (int)HID, (int)HID, 1024, 0, 0);
  // KV projection (+LN on K half, V stored transposed per (b,h))
  gemm_bf16<1><<<dim3((unsigned)(MKV / 128), 2048 / 64), 128, 0, stream>>>(
      yb, wkvT, bkv, kbuf, vT, nullptr, kn_w, kn_b, nullptr, nullptr,
      (int)MKV, (int)HID, 2048, 1024, 256, 256);
  // IP-KV projection (+LN on K half, V transposed into zero-padded buffer)
  gemm_bf16<1><<<dim3((unsigned)(MIP / 128), 2048 / 64), 128, 0, stream>>>(
      ipyb, wipT, bip, ipkbf, ipvT, nullptr, ipknw, ipknb, nullptr, nullptr,
      (int)MIP, (int)HID, 2048, 1024, 16, 32);
  // fused dual attention -> bf16 context
  attn_kernel<<<dim3(4096 / 64, 16, 8), 128, 0, stream>>>(qb, kbuf, vT, ipkbf, ipvT, ctxb);
  // output projection -> f32
  gemm_bf16<2><<<dim3((unsigned)(M1 / 128), (unsigned)(HID / 64)), 128, 0, stream>>>(
      ctxb, woT, bo, nullptr, nullptr, out, nullptr, nullptr, nullptr, nullptr,
      (int)M1, (int)HID, (int)HID, 1024, 0, 0);
}
